// LDM_17068200034721
// MI455X (gfx1250) — matrix operand order repaired and checked by reading, not compiled, not run on hardware
//
#include <hip/hip_runtime.h>
#include <hip/hip_bf16.h>
#include <math.h>

typedef float  v2f  __attribute__((ext_vector_type(2)));
typedef float  v8f  __attribute__((ext_vector_type(8)));
typedef __bf16 v4bf __attribute__((ext_vector_type(4)));
typedef __bf16 v8bf __attribute__((ext_vector_type(8)));
typedef __bf16 v16bf __attribute__((ext_vector_type(16)));

#define KDIM 128
#define TM 64          // rows per tile  (ps / n side)
#define TN 128         // cols per tile  (pp / m side)
#define LDK 136        // padded LDS row stride in bf16 elems (272B: 16B-aligned, bank-spread)
#define NB_LINK 512
#define EPS 1e-6f

// ---------------------------------------------------------------------------
// Gather + split-convert: x -> (hi = bf16(x), lo = bf16(x - hi)).
// One float4 chunk per thread; both sides handled by one grid.
// ---------------------------------------------------------------------------
__global__ __launch_bounds__(256)
void convert_kernel(const int* __restrict__ nodes_ps, const int* __restrict__ nodes_p,
                    const float* __restrict__ p_star, const float* __restrict__ p,
                    __bf16* __restrict__ Ah, __bf16* __restrict__ Al,
                    __bf16* __restrict__ Bh, __bf16* __restrict__ Bl, int B)
{
    const int idx = blockIdx.x * blockDim.x + threadIdx.x;
    const int chunksPerSide = B * (KDIM / 4);
    if (idx >= 2 * chunksPerSide) return;
    const bool sideA = idx < chunksPerSide;
    const int lidx = sideA ? idx : idx - chunksPerSide;
    const int row = lidx >> 5;            // KDIM/4 = 32 chunks per row
    const int c4  = lidx & 31;
    const int g   = sideA ? nodes_ps[row] : nodes_p[row];
    const float* src = (sideA ? p_star : p) + (size_t)g * KDIM + c4 * 4;
    const float4 v = *(const float4*)src;
    v4bf hi, lo;
    {
        const __bf16 hx = (__bf16)v.x; hi[0] = hx; lo[0] = (__bf16)(v.x - (float)hx);
        const __bf16 hy = (__bf16)v.y; hi[1] = hy; lo[1] = (__bf16)(v.y - (float)hy);
        const __bf16 hz = (__bf16)v.z; hi[2] = hz; lo[2] = (__bf16)(v.z - (float)hz);
        const __bf16 hw = (__bf16)v.w; hi[3] = hw; lo[3] = (__bf16)(v.w - (float)hw);
    }
    __bf16* dh = (sideA ? Ah : Bh) + (size_t)row * KDIM + c4 * 4;
    __bf16* dl = (sideA ? Al : Bl) + (size_t)row * KDIM + c4 * 4;
    *(v4bf*)dh = hi;
    *(v4bf*)dl = lo;
}

// ---------------------------------------------------------------------------
// Per-sample squared norms (full fp32) and gathered betas.
// ---------------------------------------------------------------------------
__global__ __launch_bounds__(256)
void norms_kernel(const int* __restrict__ nodes_ps, const int* __restrict__ nodes_p,
                  const float* __restrict__ p_star, const float* __restrict__ p,
                  const float* __restrict__ beta_ps, const float* __restrict__ beta_p,
                  float* __restrict__ aNorm, float* __restrict__ aBeta,
                  float* __restrict__ bNorm, float* __restrict__ bBeta, int N)
{
    const int lane = threadIdx.x & 31;
    const int gw   = (blockIdx.x * blockDim.x + threadIdx.x) >> 5;
    if (gw >= N) return;
    const int ia = nodes_ps[gw];
    const int ib = nodes_p[gw];
    const float4 va = ((const float4*)(p_star + (size_t)ia * KDIM))[lane];
    const float4 vb = ((const float4*)(p      + (size_t)ib * KDIM))[lane];
    float sa = va.x*va.x + va.y*va.y + va.z*va.z + va.w*va.w;
    float sb = vb.x*vb.x + vb.y*vb.y + vb.z*vb.z + vb.w*vb.w;
    for (int m = 16; m > 0; m >>= 1) {
        sa += __shfl_xor(sa, m, 32);
        sb += __shfl_xor(sb, m, 32);
    }
    if (lane == 0) {
        aNorm[gw] = sa;  aBeta[gw] = beta_ps[ia];
        bNorm[gw] = sb;  bBeta[gw] = beta_p[ib];
    }
}

// ---------------------------------------------------------------------------
// Link term: one wave per edge, grid-stride. Deterministic per-block partials.
// ---------------------------------------------------------------------------
__global__ __launch_bounds__(256)
void link_kernel(const int* __restrict__ edges, int E,
                 const float* __restrict__ beta_p, const float* __restrict__ beta_ps,
                 const float* __restrict__ p, const float* __restrict__ p_star,
                 float* __restrict__ linkPartial)
{
    const int tid  = threadIdx.x;
    const int lane = tid & 31;
    const int w    = tid >> 5;
    const int wavesTotal = gridDim.x * 8;
    float wsum = 0.0f;
    for (int e = blockIdx.x * 8 + w; e < E; e += wavesTotal) {
        const int e0 = edges[e];        // indexes p, beta_p_star
        const int e1 = edges[E + e];    // indexes p_star, beta_p
        const float4 a = ((const float4*)(p      + (size_t)e0 * KDIM))[lane];
        const float4 b = ((const float4*)(p_star + (size_t)e1 * KDIM))[lane];
        const float dx = a.x - b.x + EPS;
        const float dy = a.y - b.y + EPS;
        const float dz = a.z - b.z + EPS;
        const float dw = a.w - b.w + EPS;
        float s = dx*dx + dy*dy + dz*dz + dw*dw;
        for (int m = 16; m > 0; m >>= 1) s += __shfl_xor(s, m, 32);
        if (lane == 0)
            wsum += (beta_ps[e0] + beta_p[e1]) - sqrtf(s);
    }
    __shared__ float red[8];
    if (lane == 0) red[w] = wsum;
    __syncthreads();
    if (tid == 0) {
        float s = 0.0f;
        for (int i = 0; i < 8; ++i) s += red[i];
        linkPartial[blockIdx.x] = s;
    }
}

// ---------------------------------------------------------------------------
// Non-link term: 64x128 tile per block, split-bf16 WMMA
// (hi*hi + hi*lo + lo*hi via v_wmma_f32_16x16x32_bf16; ~fp32 accuracy).
// 8 waves arranged 4(n) x 2(m); each wave owns a 16x64 strip = 4 accumulators.
// ---------------------------------------------------------------------------
__global__ __launch_bounds__(256)
void nonlink_tile_kernel(const __bf16* __restrict__ Ah, const __bf16* __restrict__ Al,
                         const __bf16* __restrict__ Bh, const __bf16* __restrict__ Bl,
                         const float* __restrict__ aNorm, const float* __restrict__ aBeta,
                         const float* __restrict__ bNorm, const float* __restrict__ bBeta,
                         float* __restrict__ tilePartial)
{
    extern __shared__ __align__(16) char smembase[];
    __bf16* Ahs = (__bf16*)smembase;       // TM x LDK
    __bf16* Als = Ahs + TM * LDK;
    __bf16* Bhs = Als + TM * LDK;          // TN x LDK
    __bf16* Bls = Bhs + TN * LDK;

    const int tileM  = blockIdx.x;            // column tile (m / pp)
    const int tileN  = blockIdx.y;            // row tile    (n / ps)
    const int n_base = tileN * TM;
    const int m_base = tileM * TN;
    const int tid    = threadIdx.x;
    const int outIdx = tileN * gridDim.x + tileM;

    // Tile entirely at/below diagonal contributes nothing (need gn < gm).
    if (n_base >= m_base + TN - 1) {
        if (tid == 0) tilePartial[outIdx] = 0.0f;
        return;
    }

    // ---- stage pre-converted bf16 tiles into LDS (16B chunks = 8 bf16) ----
    for (int i = tid; i < TM * (KDIM / 8); i += 256) {
        const int row = i >> 4;               // 16 chunks per row
        const int c   = i & 15;
        const size_t gsrc = (size_t)(n_base + row) * KDIM + c * 8;
        *(v8bf*)(Ahs + row * LDK + c * 8) = *(const v8bf*)(Ah + gsrc);
        *(v8bf*)(Als + row * LDK + c * 8) = *(const v8bf*)(Al + gsrc);
    }
    for (int i = tid; i < TN * (KDIM / 8); i += 256) {
        const int row = i >> 4;
        const int c   = i & 15;
        const size_t gsrc = (size_t)(m_base + row) * KDIM + c * 8;
        *(v8bf*)(Bhs + row * LDK + c * 8) = *(const v8bf*)(Bh + gsrc);
        *(v8bf*)(Bls + row * LDK + c * 8) = *(const v8bf*)(Bl + gsrc);
    }
    __syncthreads();

    // ---- WMMA fragments (wave32, 16x16x32 bf16) ----
    const int w    = tid >> 5;                // 0..7
    const int lane = tid & 31;
    const int hi   = lane >> 4;
    const int r    = lane & 15;               // M for A-frag, N for B-frag / C cols
    const int n_off = (w & 3) * 16;           // wave's row strip within tile
    const int m_off = (w >> 2) * 64;          // wave's col group within tile

    v8f acc[4] = {};
    const __bf16* aHiRow = Ahs + (n_off + r) * LDK;
    const __bf16* aLoRow = Als + (n_off + r) * LDK;
    const __bf16* bHiRow = Bhs + (m_off + r) * LDK;
    const __bf16* bLoRow = Bls + (m_off + r) * LDK;

    #pragma unroll
    for (int kc = 0; kc < KDIM; kc += 32) {
        // A 16x32 bf16 layout: lane holds K in [kc+8*hi, +8) and [kc+16+8*hi, +8)
        const int ka0 = kc + 8 * hi;
        const int ka1 = kc + 16 + 8 * hi;
        const v8bf ah0 = *(const v8bf*)(aHiRow + ka0);
        const v8bf ah1 = *(const v8bf*)(aHiRow + ka1);
        const v8bf al0 = *(const v8bf*)(aLoRow + ka0);
        const v8bf al1 = *(const v8bf*)(aLoRow + ka1);
        const v16bf ahi = __builtin_shufflevector(ah0, ah1, 0,1,2,3,4,5,6,7,8,9,10,11,12,13,14,15);
        const v16bf alo = __builtin_shufflevector(al0, al1, 0,1,2,3,4,5,6,7,8,9,10,11,12,13,14,15);
        // B 32x16 bf16 layout: lane holds K in [kc+8*hi, +8) and [kc+16+8*hi, +8)
        const int kb = kc + 8 * hi;
        #pragma unroll
        for (int t = 0; t < 4; ++t) {
            const __bf16* bh = bHiRow + t * 16 * LDK + kb;
            const __bf16* bl = bLoRow + t * 16 * LDK + kb;
            const v8bf bh0 = *(const v8bf*)(bh);
            const v8bf bh1 = *(const v8bf*)(bh + 16);
            const v8bf bl0 = *(const v8bf*)(bl);
            const v8bf bl1 = *(const v8bf*)(bl + 16);
            const v16bf bhi = __builtin_shufflevector(bh0, bh1, 0,1,2,3,4,5,6,7,8,9,10,11,12,13,14,15);
            const v16bf blo = __builtin_shufflevector(bl0, bl1, 0,1,2,3,4,5,6,7,8,9,10,11,12,13,14,15);
            acc[t] = __builtin_amdgcn_wmma_f32_16x16x32_bf16(false, ahi, false, bhi, (short)0, acc[t], false, false);
            acc[t] = __builtin_amdgcn_wmma_f32_16x16x32_bf16(false, ahi, false, blo, (short)0, acc[t], false, false);
            acc[t] = __builtin_amdgcn_wmma_f32_16x16x32_bf16(false, alo, false, bhi, (short)0, acc[t], false, false);
        }
    }

    // ---- epilogue: sq -> sqrt -> exp, strict upper-triangular mask ----
    float lsum = 0.0f;
    #pragma unroll
    for (int t = 0; t < 4; ++t) {
        const int gm = m_base + m_off + t * 16 + r;        // C col = lane&15
        const float bn = bNorm[gm];
        const float bb = bBeta[gm];
        #pragma unroll
        for (int v = 0; v < 8; ++v) {
            const int gn = n_base + n_off + v + 8 * hi;    // C row = v + 8*(lane>=16)
            if (gn < gm) {
                const float s  = acc[t][v];
                const float sq = aNorm[gn] + bn - 2.0f * s;
                const float d  = sqrtf(fmaxf(sq, 0.0f));
                lsum += __expf(aBeta[gn] + bb - d);
            }
        }
    }

    // ---- deterministic block reduction ----
    for (int m = 16; m > 0; m >>= 1) lsum += __shfl_xor(lsum, m, 32);
    __syncthreads();                          // done reading LDS tiles, reuse
    float* red = (float*)smembase;
    if (lane == 0) red[w] = lsum;
    __syncthreads();
    if (tid == 0) {
        float s = 0.0f;
        for (int i = 0; i < 8; ++i) s += red[i];
        tilePartial[outIdx] = s;
    }
}

// ---------------------------------------------------------------------------
// Final deterministic reduction: out = non_link - link_term.
// ---------------------------------------------------------------------------
__global__ __launch_bounds__(256)
void finalize_kernel(const float* __restrict__ linkPartial, int nLink,
                     const float* __restrict__ tilePartial, int nTile,
                     float* __restrict__ out)
{
    __shared__ float red[256];
    const int tid = threadIdx.x;
    float s = 0.0f;
    for (int i = tid; i < nTile; i += 256) s += tilePartial[i];
    float l = 0.0f;
    for (int i = tid; i < nLink; i += 256) l += linkPartial[i];
    red[tid] = s - l;
    __syncthreads();
    for (int st = 128; st > 0; st >>= 1) {
        if (tid < st) red[tid] += red[tid + st];
        __syncthreads();
    }
    if (tid == 0) out[0] = red[0];
}

// ---------------------------------------------------------------------------
extern "C" void kernel_launch(void* const* d_in, const int* in_sizes, int n_in,
                              void* d_out, int out_size, void* d_ws, size_t ws_size,
                              hipStream_t stream)
{
    const int*   edges    = (const int*)d_in[0];
    const int*   nodes_ps = (const int*)d_in[1];   // nodes_p_star
    const int*   nodes_p  = (const int*)d_in[2];
    const float* beta_p   = (const float*)d_in[3];
    const float* beta_ps  = (const float*)d_in[4]; // beta_p_star
    const float* p        = (const float*)d_in[5];
    const float* p_star   = (const float*)d_in[6];
    float* out = (float*)d_out;

    const int E = in_sizes[0] / 2;     // edges shape (2, E)
    const int B = in_sizes[1];         // 16384 sampled nodes per side

    // workspace layout: 4 bf16 matrices (hi/lo per side), then fp32 scratch
    __bf16* Ah = (__bf16*)d_ws;                    // [B*KDIM]
    __bf16* Al = Ah + (size_t)B * KDIM;
    __bf16* Bh = Al + (size_t)B * KDIM;
    __bf16* Bl = Bh + (size_t)B * KDIM;
    float* linkPartial = (float*)(Bl + (size_t)B * KDIM);   // [NB_LINK]
    float* aNorm       = linkPartial + NB_LINK;    // [B]
    float* aBeta       = aNorm + B;                // [B]
    float* bNorm       = aBeta + B;                // [B]
    float* bBeta       = bNorm + B;                // [B]
    float* tilePartial = bBeta + B;                // [ntN * ntM]

    const int ntM = B / TN;            // 128 column tiles
    const int ntN = B / TM;            // 256 row tiles

    convert_kernel<<<(2 * B * (KDIM / 4) + 255) / 256, 256, 0, stream>>>(
        nodes_ps, nodes_p, p_star, p, Ah, Al, Bh, Bl, B);

    norms_kernel<<<(B * 32 + 255) / 256, 256, 0, stream>>>(
        nodes_ps, nodes_p, p_star, p, beta_ps, beta_p,
        aNorm, aBeta, bNorm, bBeta, B);

    link_kernel<<<NB_LINK, 256, 0, stream>>>(
        edges, E, beta_p, beta_ps, p, p_star, linkPartial);

    dim3 grid(ntM, ntN);
    const size_t ldsBytes = (size_t)(2 * TM * LDK + 2 * TN * LDK) * sizeof(__bf16); // ~102 KB
    nonlink_tile_kernel<<<grid, 256, ldsBytes, stream>>>(
        Ah, Al, Bh, Bl, aNorm, aBeta, bNorm, bBeta, tilePartial);

    finalize_kernel<<<1, 256, 0, stream>>>(
        linkPartial, NB_LINK, tilePartial, ntM * ntN, out);
}
